// GnS_31662498906135
// MI455X (gfx1250) — matrix operand order repaired and checked by reading, not compiled, not run on hardware
//
#include <hip/hip_runtime.h>

// ---------------- types for WMMA ----------------
typedef __attribute__((ext_vector_type(16))) __bf16 v16bf;
typedef __attribute__((ext_vector_type(8)))  float  v8f;

union Frag {
    unsigned int u[8];
    v16bf v;
};

__device__ __forceinline__ unsigned short f2bf(float f) {
    unsigned int u = __builtin_bit_cast(unsigned int, f);
    unsigned int r = u + 0x7FFFu + ((u >> 16) & 1u);   // RNE
    return (unsigned short)(r >> 16);
}
__device__ __forceinline__ float bf2f(unsigned short h) {
    unsigned int u = ((unsigned int)h) << 16;
    return __builtin_bit_cast(float, u);
}

// ---------------- constants ----------------
#define NN 1000000
#define EE 4000000
#define BB 1024
#define FX 55
#define DIM 32
#define LL 1000
#define EMBD 128
#define CONV_FLAT 3872   // 32*121
#define HOUT 121

// ---------------- elementwise / util kernels ----------------
__global__ void zero_f32(float* p, long long n) {
    for (long long i = blockIdx.x * (long long)blockDim.x + threadIdx.x; i < n;
         i += (long long)gridDim.x * blockDim.x)
        p[i] = 0.0f;
}

// f32 [rows, cols] -> bf16 [rowsPad, cols], zero padding rows >= rows
__global__ void pad_to_bf16(const float* __restrict__ src, unsigned short* __restrict__ dst,
                            int rows, int cols, int rowsPad) {
    long long total = (long long)rowsPad * cols;
    for (long long i = blockIdx.x * (long long)blockDim.x + threadIdx.x; i < total;
         i += (long long)gridDim.x * blockDim.x) {
        long long r = i / cols;
        int c = (int)(i - r * cols);
        float v = (r < rows) ? src[r * cols + c] : 0.0f;
        dst[i] = f2bf(v);
    }
}

// conv_k [32,1000,8] f32  ->  W [8000,32] bf16  (W[(i*8+k)*32 + o])
__global__ void conv_w_transpose(const float* __restrict__ src, unsigned short* __restrict__ dst) {
    long long total = 8000LL * 32;
    for (long long i = blockIdx.x * (long long)blockDim.x + threadIdx.x; i < total;
         i += (long long)gridDim.x * blockDim.x) {
        int kk = (int)(i >> 5);
        int o  = (int)(i & 31);
        dst[i] = f2bf(src[(long long)o * 8000 + kk]);
    }
}

// edge scatter-add: agg[dst] += x[src]   (f32 atomics; node features stay L2-resident)
__global__ __launch_bounds__(256) void scatter_agg(const float* __restrict__ x,
                                                   const int* __restrict__ src,
                                                   const int* __restrict__ dst,
                                                   float* __restrict__ agg, int D) {
    int e = blockIdx.x * blockDim.x + threadIdx.x;
    if (e >= EE) return;
    long long s = (long long)src[e] * D;
    long long d = (long long)dst[e] * D;
    for (int c = 0; c < D; ++c)
        atomicAdd(&agg[d + c], x[s + c]);
}

// abf[n, c] = bf16( x[n,c] + agg[n,c] ) with zero-pad to KP columns
__global__ void stage_xin(const float* __restrict__ x, const float* __restrict__ agg,
                          unsigned short* __restrict__ abf, int D, int KP) {
    long long total = (long long)NN * KP;
    for (long long i = blockIdx.x * (long long)blockDim.x + threadIdx.x; i < total;
         i += (long long)gridDim.x * blockDim.x) {
        int c = (int)(i % KP);
        long long r = i / KP;
        float v = 0.0f;
        if (c < D) {
            long long o = r * D + c;
            v = x[o] + agg[o];
        }
        abf[i] = f2bf(v);
    }
}

// per-column sum / sumsq over h [NN,32] -> sums[0..31]=sum, sums[32..63]=sumsq
__global__ __launch_bounds__(256) void col_stats(const float* __restrict__ h, float* __restrict__ sums) {
    __shared__ float s1[256], s2[256];
    int c = threadIdx.x & 31;
    int wavesPerBlock = blockDim.x >> 5;
    float a = 0.f, b = 0.f;
    for (long long r = blockIdx.x * (long long)wavesPerBlock + (threadIdx.x >> 5); r < NN;
         r += (long long)gridDim.x * wavesPerBlock) {
        float v = h[r * 32 + c];
        a += v; b += v * v;
    }
    s1[threadIdx.x] = a; s2[threadIdx.x] = b;
    __syncthreads();
    if (threadIdx.x < 32) {
        for (int w = 1; w < wavesPerBlock; ++w) { a += s1[w * 32 + c]; b += s2[w * 32 + c]; }
        atomicAdd(&sums[c], a);
        atomicAdd(&sums[32 + c], b);
    }
}

// in-place batchnorm using batch stats + affine
__global__ void bn_apply(float* __restrict__ h, const float* __restrict__ sums,
                         const float* __restrict__ g, const float* __restrict__ bt) {
    long long total = (long long)NN * 32;
    float invN = 1.0f / (float)NN;
    for (long long i = blockIdx.x * (long long)blockDim.x + threadIdx.x; i < total;
         i += (long long)gridDim.x * blockDim.x) {
        int c = (int)(i & 31);
        float m = sums[c] * invN;
        float var = sums[32 + c] * invN - m * m;
        float inv = rsqrtf(var + 1e-5f);
        h[i] = (h[i] - m) * inv * g[c] + bt[c];
    }
}

// pooled[batch[n]] += h[n]
__global__ __launch_bounds__(256) void pool_scatter(const float* __restrict__ h,
                                                    const int* __restrict__ batch,
                                                    float* __restrict__ pooled) {
    int n = blockIdx.x * blockDim.x + threadIdx.x;
    if (n >= NN) return;
    long long b = (long long)batch[n] * 32;
    long long o = (long long)n * 32;
    for (int c = 0; c < 32; ++c)
        atomicAdd(&pooled[b + c], h[o + c]);
}

// ebf[(b*1000+i)*128 + c] = bf16(emb[xt[b*1000+i]][c])
__global__ __launch_bounds__(256) void stage_emb(const int* __restrict__ xt,
                                                 const float* __restrict__ emb,
                                                 unsigned short* __restrict__ ebf) {
    long long t = blockIdx.x * (long long)blockDim.x + threadIdx.x;
    if (t >= (long long)BB * LL) return;
    const float* er = emb + (long long)xt[t] * EMBD;
    unsigned short* out = ebf + t * EMBD;
    for (int c = 0; c < EMBD; ++c)
        out[c] = f2bf(er[c]);
}

// ---------------- weight-stationary GIN GEMM ----------------
// C[NN,32] = relu( A[NN, KSLABS*32] * Bw[KSLABS*32, 32] + bias )
// B fragments live in registers for the whole wave; each A fragment feeds
// 2 WMMAs (both N-tiles). Grid-stride over M-tiles.
template<int KSLABS, bool WRITE_F32>
__global__ __launch_bounds__(256) void gin_gemm(
    const unsigned short* __restrict__ A,
    const unsigned short* __restrict__ Bw,
    const float* __restrict__ bias,
    float* __restrict__ Cf,
    unsigned short* __restrict__ Cb)
{
    const int lda = KSLABS * 32;
    int lane = threadIdx.x & 31;
    int rlo = lane & 15, hi = lane >> 4;

    // hoist B fragments (weight-stationary): KSLABS k-slabs x 2 n-tiles
    Frag bfr[KSLABS][2];
#pragma unroll
    for (int s = 0; s < KSLABS; ++s)
#pragma unroll
        for (int t = 0; t < 2; ++t)
#pragma unroll
            for (int v = 0; v < 8; ++v) {
                int kb = s * 32 + 2 * v + (hi ? 16 : 0);
                int col = t * 16 + rlo;
                unsigned int blo = Bw[kb * 32 + col];
                unsigned int bhi = Bw[(kb + 1) * 32 + col];
                bfr[s][t].u[v] = blo | (bhi << 16);
            }
    float bc0 = bias[rlo];
    float bc1 = bias[16 + rlo];

    int wave = blockIdx.x * (blockDim.x >> 5) + (threadIdx.x >> 5);
    int nwaves = gridDim.x * (blockDim.x >> 5);
    for (int tm = wave; tm < NN / 16; tm += nwaves) {
        long long rowoff = (long long)(tm * 16 + rlo) * lda;
        v8f acc0 = {}, acc1 = {};
#pragma unroll
        for (int s = 0; s < KSLABS; ++s) {
            Frag a;
#pragma unroll
            for (int v = 0; v < 8; ++v) {
                int ka = s * 32 + 2 * v + (hi ? 16 : 0);
                a.u[v] = *(const unsigned int*)(A + rowoff + ka);
            }
            acc0 = __builtin_amdgcn_wmma_f32_16x16x32_bf16(false, a.v, false, bfr[s][0].v,
                                                           (short)0, acc0, false, false);
            acc1 = __builtin_amdgcn_wmma_f32_16x16x32_bf16(false, a.v, false, bfr[s][1].v,
                                                           (short)0, acc1, false, false);
        }
#pragma unroll
        for (int v = 0; v < 8; ++v) {
            int m = tm * 16 + v + (hi ? 8 : 0);
            float x0 = fmaxf(acc0[v] + bc0, 0.0f);
            float x1 = fmaxf(acc1[v] + bc1, 0.0f);
            if (WRITE_F32) {
                Cf[(long long)m * 32 + rlo] = x0;
                Cf[(long long)m * 32 + 16 + rlo] = x1;
            } else {
                Cb[(long long)m * 32 + rlo] = f2bf(x0);
                Cb[(long long)m * 32 + 16 + rlo] = f2bf(x1);
            }
        }
    }
}

// ---------------- generic WMMA GEMM (bf16 in/out, f32 accum) ----------------
// Cb[M,N] = bf16( relu?( A[M,K] * B[K,N] + bias[N] ) )
// Requires: M%16==0, N%16==0, K%32==0, lda even.
template<bool RELU>
__global__ __launch_bounds__(128) void wmma_gemm_bf16(
    const unsigned short* __restrict__ A, int lda,
    const unsigned short* __restrict__ Bm, int ldb,
    const float* __restrict__ bias,
    unsigned short* __restrict__ Cb, int ldcb,
    int M, int N, int K)
{
    int lane = threadIdx.x & 31;
    int wave = blockIdx.x * (blockDim.x >> 5) + (threadIdx.x >> 5);
    int ntiles = N >> 4;
    int total = (M >> 4) * ntiles;
    if (wave >= total) return;
    int tm = wave / ntiles, tn = wave - tm * ntiles;
    int rlo = lane & 15;
    int hi  = lane >> 4;
    int row = tm * 16 + rlo;
    int col = tn * 16 + rlo;
    const unsigned short* Arow = A + (long long)row * lda;

    v8f acc = {};
    for (int k0 = 0; k0 < K; k0 += 32) {
        __builtin_prefetch(Arow + k0 + 128, 0, 0);   // stream A ahead (global_prefetch_b8)
        Frag a, b;
#pragma unroll
        for (int v = 0; v < 8; ++v) {
            int ka = k0 + 2 * v + (hi ? 16 : 0);
            a.u[v] = *(const unsigned int*)(Arow + ka);
            int kb = k0 + 2 * v + (hi ? 16 : 0);
            unsigned int blo = Bm[(long long)kb * ldb + col];
            unsigned int bhi = Bm[(long long)(kb + 1) * ldb + col];
            b.u[v] = blo | (bhi << 16);
        }
        acc = __builtin_amdgcn_wmma_f32_16x16x32_bf16(false, a.v, false, b.v,
                                                      (short)0, acc, false, false);
    }

    float bc = bias[col];
#pragma unroll
    for (int v = 0; v < 8; ++v) {
        int m = tm * 16 + v + (hi ? 8 : 0);
        float x = acc[v] + bc;
        if (RELU) x = fmaxf(x, 0.0f);
        Cb[(long long)m * ldcb + col] = f2bf(x);
    }
}

// ---------------- conv as 121 gathered-A WMMA GEMMs ----------------
// out[b,o,h'] = sum_{i,k} ebf[b,i,h'+k] * W[(i*8+k), o] + bias[o]
// One wave per (h', M-tile): gathers one A fragment per K-slab and feeds both
// output-channel N-tiles (2 WMMAs / slab). Stored bf16 at Cb[b*3872 + o*121 + h'].
__global__ __launch_bounds__(128) void conv_wmma(
    const unsigned short* __restrict__ Ebf,   // [B,1000,128] bf16
    const unsigned short* __restrict__ Wb,    // [8000,32] bf16
    const float* __restrict__ bias,           // [32]
    unsigned short* __restrict__ Cb)          // [B,3872] bf16
{
    int lane = threadIdx.x & 31;
    int wave = blockIdx.x * (blockDim.x >> 5) + (threadIdx.x >> 5);
    const int total = HOUT * (BB / 16);       // 121 * 64
    if (wave >= total) return;
    int hp = wave % HOUT;
    int tm = wave / HOUT;
    int rlo = lane & 15;
    int hi  = lane >> 4;
    int brow = tm * 16 + rlo;
    const unsigned short* Arow = Ebf + (long long)brow * (LL * EMBD) + hp;

    v8f acc0 = {}, acc1 = {};
    for (int i0 = 0; i0 < LL; i0 += 4) {      // K slab of 32 = 4 positions x 8 taps
        __builtin_prefetch(Arow + (long long)(i0 + 16) * EMBD, 0, 0);
        int kbase = i0 * 8;
        Frag a, b0, b1;
#pragma unroll
        for (int v = 0; v < 8; ++v) {
            int kk = 2 * v + (hi ? 16 : 0); // 0..31
            int i = i0 + (kk >> 3);
            int k = kk & 7;
            const unsigned short* p = Arow + (long long)i * EMBD + k;    // pair contiguous
            a.u[v] = (unsigned int)p[0] | ((unsigned int)p[1] << 16);
            int kb = kbase + 2 * v + (hi ? 16 : 0);
            const unsigned short* q0 = Wb + (long long)kb * 32;
            b0.u[v] = (unsigned int)q0[rlo]      | ((unsigned int)q0[32 + rlo]      << 16);
            b1.u[v] = (unsigned int)q0[16 + rlo] | ((unsigned int)q0[32 + 16 + rlo] << 16);
        }
        acc0 = __builtin_amdgcn_wmma_f32_16x16x32_bf16(false, a.v, false, b0.v,
                                                       (short)0, acc0, false, false);
        acc1 = __builtin_amdgcn_wmma_f32_16x16x32_bf16(false, a.v, false, b1.v,
                                                       (short)0, acc1, false, false);
    }

    float bc0 = bias[rlo];
    float bc1 = bias[16 + rlo];
#pragma unroll
    for (int v = 0; v < 8; ++v) {
        int m = tm * 16 + v + (hi ? 8 : 0);
        Cb[(long long)m * CONV_FLAT + rlo * HOUT + hp]        = f2bf(acc0[v] + bc0);
        Cb[(long long)m * CONV_FLAT + (16 + rlo) * HOUT + hp] = f2bf(acc1[v] + bc1);
    }
}

// ---------------- final projection + y passthrough ----------------
__global__ __launch_bounds__(256) void cls3_out(const unsigned short* __restrict__ h2,
                                                const float* __restrict__ w3,
                                                const float* __restrict__ b3,
                                                const float* __restrict__ y,
                                                float* __restrict__ out) {
    int wave = blockIdx.x * (blockDim.x >> 5) + (threadIdx.x >> 5);
    if (wave >= BB) return;
    int lane = threadIdx.x & 31;
    float s = 0.0f;
    for (int j = lane; j < 256; j += 32)
        s += bf2f(h2[(long long)wave * 256 + j]) * w3[j];
    for (int off = 16; off > 0; off >>= 1)
        s += __shfl_down(s, off, 32);
    if (lane == 0) {
        out[wave] = s + b3[0];
        out[BB + wave] = y[wave];
    }
}

// ---------------- host-side helpers ----------------
template<bool RELU>
static inline void launch_gemm(hipStream_t s, const unsigned short* A, int lda,
                               const unsigned short* B, int ldb, const float* bias,
                               unsigned short* Cb, int ldcb, int M, int N, int K) {
    int tiles = (M / 16) * (N / 16);
    int blocks = (tiles + 3) / 4;
    wmma_gemm_bf16<RELU><<<blocks, 128, 0, s>>>(A, lda, B, ldb, bias, Cb, ldcb, M, N, K);
}

extern "C" void kernel_launch(void* const* d_in, const int* in_sizes, int n_in,
                              void* d_out, int out_size, void* d_ws, size_t ws_size,
                              hipStream_t stream) {
    (void)in_sizes; (void)n_in; (void)out_size; (void)ws_size;
    // ---- inputs ----
    const float* xd      = (const float*)d_in[0];
    const int*   ei      = (const int*)d_in[1];
    const int*   batch   = (const int*)d_in[2];
    const int*   xt      = (const int*)d_in[3];
    const float* y       = (const float*)d_in[4];
    const float* gin1_w1 = (const float*)d_in[5];
    const float* gin1_b1 = (const float*)d_in[6];
    const float* gin1_w2 = (const float*)d_in[7];
    const float* gin1_b2 = (const float*)d_in[8];
    const float* ginr_w1 = (const float*)d_in[9];
    const float* ginr_b1 = (const float*)d_in[10];
    const float* ginr_w2 = (const float*)d_in[11];
    const float* ginr_b2 = (const float*)d_in[12];
    const float* bn_g    = (const float*)d_in[13];
    const float* bn_b    = (const float*)d_in[14];
    const float* fcd_w   = (const float*)d_in[15];
    const float* fcd_b   = (const float*)d_in[16];
    const float* emb     = (const float*)d_in[17];
    const float* conv_k  = (const float*)d_in[18];
    const float* conv_b  = (const float*)d_in[19];
    const float* fct_w   = (const float*)d_in[20];
    const float* fct_b   = (const float*)d_in[21];
    const float* cls_w1  = (const float*)d_in[22];
    const float* cls_b1  = (const float*)d_in[23];
    const float* cls_w2  = (const float*)d_in[24];
    const float* cls_b2  = (const float*)d_in[25];
    const float* cls_w3  = (const float*)d_in[26];
    const float* cls_b3  = (const float*)d_in[27];
    const int* src = ei;
    const int* dst = ei + EE;
    float* out = (float*)d_out;

    // ---- workspace layout (phase-aliased) ----
    char* ws = (char*)d_ws;
    size_t off = 0;
    // R1: agg (graph phase, <=220MB) aliased with ebf (text phase, 262.1MB)
    float*          agg   = (float*)(ws + off);
    unsigned short* ebf   = (unsigned short*)(ws + off);      off += 262144000;
    // R2: abf (graph phase, 128MB) aliased with cbf (text phase, 7.9MB)
    unsigned short* abf   = (unsigned short*)(ws + off);
    unsigned short* cbf   = (unsigned short*)(ws + off);      off += 128000000;
    // R3: h1bf (graph phase, 64MB) aliased with classifier hidden (2MB)
    unsigned short* h1bf  = (unsigned short*)(ws + off);
    unsigned short* ch1bf = (unsigned short*)(ws + off);      off += 64000000;
    // R4: node features f32 (L2-resident 128MB)
    float*          hbuf  = (float*)(ws + off);               off += 128000000;
    // smalls
    float*          sums  = (float*)(ws + off);               off += 256;
    float*          pooled= (float*)(ws + off);               off += 131072;
    unsigned short* pbf   = (unsigned short*)(ws + off);      off += 65536;
    unsigned short* xjbf  = (unsigned short*)(ws + off);      off += 524288;
    unsigned short* h2bf  = (unsigned short*)(ws + off);      off += 524288;
    unsigned short* wa_bf = (unsigned short*)(ws + off);      off += 4096;    // up to 64x32
    unsigned short* wb_bf = (unsigned short*)(ws + off);      off += 2048;    // 32x32
    unsigned short* fcdwb = (unsigned short*)(ws + off);      off += 8192;
    unsigned short* fctwb = (unsigned short*)(ws + off);      off += 991232;
    unsigned short* c1wb  = (unsigned short*)(ws + off);      off += 524288;
    unsigned short* c2wb  = (unsigned short*)(ws + off);      off += 524288;
    unsigned short* wconv = (unsigned short*)(ws + off);      off += 512000;

    const int GS = 2048;  // grid for grid-stride elementwise kernels

    // ---- one-time (per launch) weight staging ----
    pad_to_bf16<<<GS, 256, 0, stream>>>(fcd_w, fcdwb, 32, 128, 32);
    pad_to_bf16<<<GS, 256, 0, stream>>>(fct_w, fctwb, CONV_FLAT, 128, CONV_FLAT);
    pad_to_bf16<<<GS, 256, 0, stream>>>(cls_w1, c1wb, 256, 1024, 256);
    pad_to_bf16<<<GS, 256, 0, stream>>>(cls_w2, c2wb, 1024, 256, 1024);
    conv_w_transpose<<<GS, 256, 0, stream>>>(conv_k, wconv);

    // ---- graph branch: 5 GIN layers ----
    const float* x = xd;
    for (int layer = 0; layer < 5; ++layer) {
        int D  = (layer == 0) ? FX : DIM;   // 55 or 32
        int KP = (layer == 0) ? 64 : 32;    // K padded to multiple of 32
        const float* w1 = (layer == 0) ? gin1_w1 : (ginr_w1 + (layer - 1) * DIM * DIM);
        const float* b1 = (layer == 0) ? gin1_b1 : (ginr_b1 + (layer - 1) * DIM);
        const float* w2 = (layer == 0) ? gin1_w2 : (ginr_w2 + (layer - 1) * DIM * DIM);
        const float* b2 = (layer == 0) ? gin1_b2 : (ginr_b2 + (layer - 1) * DIM);

        zero_f32<<<GS, 256, 0, stream>>>(agg, (long long)NN * D);
        scatter_agg<<<(EE + 255) / 256, 256, 0, stream>>>(x, src, dst, agg, D);
        stage_xin<<<GS, 256, 0, stream>>>(x, agg, abf, D, KP);
        pad_to_bf16<<<GS, 256, 0, stream>>>(w1, wa_bf, D, DIM, KP);
        pad_to_bf16<<<GS, 256, 0, stream>>>(w2, wb_bf, DIM, DIM, DIM);

        // h1 = relu((x+agg) @ w1 + b1)  -> bf16  (weight-stationary WMMA)
        if (layer == 0)
            gin_gemm<2, false><<<2048, 256, 0, stream>>>(abf, wa_bf, b1, nullptr, h1bf);
        else
            gin_gemm<1, false><<<2048, 256, 0, stream>>>(abf, wa_bf, b1, nullptr, h1bf);
        // h = relu(h1 @ w2 + b2)        -> f32
        gin_gemm<1, true><<<2048, 256, 0, stream>>>(h1bf, wb_bf, b2, hbuf, nullptr);

        zero_f32<<<1, 64, 0, stream>>>(sums, 64);
        col_stats<<<1024, 256, 0, stream>>>(hbuf, sums);
        bn_apply<<<GS, 256, 0, stream>>>(hbuf, sums, bn_g + layer * DIM, bn_b + layer * DIM);
        x = hbuf;
    }

    // ---- pooling + fcd -> xj[:, 0:128] ----
    zero_f32<<<GS, 256, 0, stream>>>(pooled, (long long)BB * DIM);
    pool_scatter<<<(NN + 255) / 256, 256, 0, stream>>>(hbuf, batch, pooled);
    pad_to_bf16<<<GS, 256, 0, stream>>>(pooled, pbf, BB, DIM, BB);
    launch_gemm<true>(stream, pbf, DIM, fcdwb, 128, fcd_b, xjbf, 256, BB, 128, DIM);

    // ---- text branch: embed -> conv(WMMA) -> fct -> xj[:, 128:256] ----
    stage_emb<<<(BB * LL + 255) / 256, 256, 0, stream>>>(xt, emb, ebf);
    {
        int waves = HOUT * (BB / 16);
        conv_wmma<<<(waves + 3) / 4, 128, 0, stream>>>(ebf, wconv, conv_b, cbf);
    }
    launch_gemm<false>(stream, cbf, CONV_FLAT, fctwb, 128, fct_b, xjbf + 128, 256,
                       BB, 128, CONV_FLAT);

    // ---- classifier ----
    launch_gemm<true>(stream, xjbf, 256, c1wb, 1024, cls_b1, ch1bf, 1024, BB, 1024, 256);
    launch_gemm<true>(stream, ch1bf, 1024, c2wb, 256, cls_b2, h2bf, 256, BB, 256, 1024);
    cls3_out<<<(BB + 7) / 8, 256, 0, stream>>>(h2bf, cls_w3, cls_b3, y, out);
}
